// StructCmap_10376640987595
// MI455X (gfx1250) — compile-verified, matrix operand order checked by reading, not run on hardware
//
#include <hip/hip_runtime.h>
#include <math.h>

typedef __attribute__((ext_vector_type(16))) __bf16 v16bf;
typedef __attribute__((ext_vector_type(8)))  float  v8f;

#define GQ  __attribute__((address_space(1)))
#define GQ3 __attribute__((address_space(3)))
typedef const float GQ* gcf;
typedef float GQ*       gf;
typedef const __bf16 GQ* gcbf;
typedef __bf16 GQ*       gbf;

// ---------------------------------------------------------------------------
// CDNA5 WMMA helpers (wave32). 16x16x32 bf16 -> f32.
// A (16x32) / B (32x16 as B^T rows): lane L: half = L>>4, row = L&15.
// element e of v16bf: v = e>>1, k = (v>=4?16:0) + half*8 + (v&3)*2 + (e&1).
// C (16x16 f32, 8 VGPRs): elem r -> M = r + 8*half, N = lane&15.
// ---------------------------------------------------------------------------
__device__ __forceinline__ v8f wmma_bf16(v16bf a, v16bf b, v8f c) {
  return __builtin_amdgcn_wmma_f32_16x16x32_bf16(false, a, false, b, (short)0, c,
                                                 false, false);
}

__device__ __forceinline__ v16bf frag_from_lds(const __bf16* buf, int ld,
                                               int rowBase, int kBase, int lane) {
  const int half = lane >> 4;
  const int row  = rowBase + (lane & 15);
  const __bf16* p = buf + row * ld + kBase + half * 8;
  v16bf f;
#pragma unroll
  for (int e = 0; e < 16; ++e) {
    const int v = e >> 1;
    const int k = ((v >= 4) ? 16 : 0) + (v & 3) * 2 + (e & 1);
    f[e] = p[k];
  }
  return f;
}

// LDS byte address (for async-to-LDS instructions)
__device__ __forceinline__ unsigned lds_addr(const void* p) {
  return (unsigned)(uintptr_t)(const GQ3 void*)p;
}

// Async 16-byte global->LDS copy (per lane), tracked by ASYNCcnt.
__device__ __forceinline__ void async_ld_b128(unsigned lds, const void* gaddr) {
  asm volatile("global_load_async_to_lds_b128 %0, %1, off"
               :
               : "v"(lds), "v"(gaddr)
               : "memory");
}
__device__ __forceinline__ void wait_asynccnt0() {
  asm volatile("s_wait_asynccnt 0x0" ::: "memory");
}

// ---------------------------------------------------------------------------
// Generic 64x64-tile GEMM: C[m,n] = act( sum_k A[m,k]*B(k,n) + bias + resid )
//   bNT=1: B(k,n) = Bsrc[n*ldb + k]; bNT=0: B(k,n) = Bsrc[k*ldb + n]
// epi: 0 = f32 (acc+bias); 1 = f32 tanh(acc+bias+resid); 2 = bf16 tanh(acc)
// 256 threads = 8 waves; wave (wy,wx) owns (wy*32, wx*16) and (wy*32+16, wx*16)
// -> B fragment reused for both WMMAs.
// ---------------------------------------------------------------------------
#define LDS_K 34

__device__ void gemm_block64(const float* __restrict__ A, int lda,
                             const float* __restrict__ Bsrc, int ldb, int K,
                             int bNT, void* Cptr, int ldc,
                             const float* __restrict__ bias,
                             const float* __restrict__ resid, int epi,
                             int m0, int n0) {
  __shared__ __bf16 As[64 * LDS_K];
  __shared__ __bf16 Bs[64 * LDS_K];
  const gcf gA = (gcf)A;
  const gcf gB = (gcf)Bsrc;
  const int tid  = threadIdx.x;
  const int lane = tid & 31;
  const int wave = tid >> 5;
  const int wx = wave & 3, wy = wave >> 2;
  const int mS0 = wy * 32, mS1 = mS0 + 16, nS = wx * 16;

  v8f c0 = {}, c1 = {};
  const int rr = tid >> 2;          // 0..63
  const int kk = (tid & 3) * 8;     // 0,8,16,24

  for (int k0 = 0; k0 < K; k0 += 32) {
    {  // stage A 64x32 (f32 -> bf16)
      const gcf src = gA + (size_t)(m0 + rr) * lda + k0 + kk;
      __bf16* dst = As + rr * LDS_K + kk;
#pragma unroll
      for (int u = 0; u < 8; ++u) dst[u] = (__bf16)src[u];
    }
    {  // stage B as [n][k]
      __bf16* dst = Bs + rr * LDS_K + kk;
      if (bNT) {
        const gcf src = gB + (size_t)(n0 + rr) * ldb + k0 + kk;
#pragma unroll
        for (int u = 0; u < 8; ++u) dst[u] = (__bf16)src[u];
      } else {
        const gcf src = gB + (size_t)(k0 + kk) * ldb + n0 + rr;
#pragma unroll
        for (int u = 0; u < 8; ++u) dst[u] = (__bf16)src[(size_t)u * ldb];
      }
    }
    __syncthreads();
    v16bf bf = frag_from_lds(Bs, LDS_K, nS, 0, lane);
    v16bf a0 = frag_from_lds(As, LDS_K, mS0, 0, lane);
    v16bf a1 = frag_from_lds(As, LDS_K, mS1, 0, lane);
    c0 = wmma_bf16(a0, bf, c0);
    c1 = wmma_bf16(a1, bf, c1);
    __syncthreads();
  }

  const int half = lane >> 4;
  const int gn = n0 + nS + (lane & 15);
#pragma unroll
  for (int r = 0; r < 8; ++r) {
    const int gmA = m0 + mS0 + r + 8 * half;
    const int gmB = m0 + mS1 + r + 8 * half;
    float vA = c0[r], vB = c1[r];
    if (bias) {
      const float bn = ((gcf)bias)[gn];
      vA += bn;
      vB += bn;
    }
    if (epi == 0) {
      ((gf)Cptr)[(size_t)gmA * ldc + gn] = vA;
      ((gf)Cptr)[(size_t)gmB * ldc + gn] = vB;
    } else if (epi == 1) {
      ((gf)Cptr)[(size_t)gmA * ldc + gn] =
          tanhf(vA + ((gcf)resid)[(size_t)gmA * ldc + gn]);
      ((gf)Cptr)[(size_t)gmB * ldc + gn] =
          tanhf(vB + ((gcf)resid)[(size_t)gmB * ldc + gn]);
    } else {
      ((gbf)Cptr)[(size_t)gmA * ldc + gn] = (__bf16)tanhf(vA);
      ((gbf)Cptr)[(size_t)gmB * ldc + gn] = (__bf16)tanhf(vB);
    }
  }
}

__global__ __launch_bounds__(256) void gemm_kernel(
    const float* A, int lda, const float* B, int ldb, int K, int bNT, float* C,
    int ldc, const float* bias, const float* resid, int epi) {
  gemm_block64(A, lda, B, ldb, K, bNT, C, ldc, bias, resid, epi,
               blockIdx.y * 64, blockIdx.x * 64);
}

// h[bb][n] = tanh( xa_bb (1024x256) @ Wbin[n] (256x128) ), stored bf16
__global__ __launch_bounds__(256) void wbin_kernel(const float* xa,
                                                   const float* Wbin,
                                                   __bf16* h) {
  const int z = blockIdx.z;
  const int n = z & 15;
  const int bb = z >> 4;  // br*2 + b
  gemm_block64(xa + (size_t)bb * 1024 * 256, 256,
               Wbin + (size_t)n * 256 * 128, 128, 256, /*bNT=*/0,
               h + ((size_t)bb * 16 + n) * 131072, 128, nullptr, nullptr,
               /*epi=*/2, blockIdx.y * 64, blockIdx.x * 64);
}

// ---------------------------------------------------------------------------
// Degenerate MHA: attention over S=2. One wave per (branch, l, head).
// ---------------------------------------------------------------------------
__device__ __forceinline__ float wave_sum(float x) {
#pragma unroll
  for (int m = 16; m >= 1; m >>= 1) x += __shfl_xor(x, m, 32);
  return x;
}

__global__ __launch_bounds__(256) void attn_kernel(const float* __restrict__ qkv,
                                                   float* __restrict__ o) {
  const gcf q = (gcf)qkv;
  gf go = (gf)o;
  const int gid  = blockIdx.x * 8 + (threadIdx.x >> 5);
  const int lane = threadIdx.x & 31;
  const int hh = gid & 7;
  const int n  = (gid >> 3) & 1023;
  const int br = gid >> 13;
  const size_t row0 = (size_t)br * 2048 + n;   // s = 0
  const size_t row1 = row0 + 1024;             // s = 1
  const int cb = hh * 32 + lane;
  const float q0 = q[row0 * 768 + cb],       q1 = q[row1 * 768 + cb];
  const float k0 = q[row0 * 768 + 256 + cb], k1 = q[row1 * 768 + 256 + cb];
  const float v0 = q[row0 * 768 + 512 + cb], v1 = q[row1 * 768 + 512 + cb];
  const float sc = 0.17677669529663687f;  // 32^-0.5
  const float d00 = wave_sum(q0 * k0) * sc;
  const float d01 = wave_sum(q0 * k1) * sc;
  const float d10 = wave_sum(q1 * k0) * sc;
  const float d11 = wave_sum(q1 * k1) * sc;
  const float m0 = fmaxf(d00, d01), e00 = expf(d00 - m0), e01 = expf(d01 - m0);
  const float m1 = fmaxf(d10, d11), e10 = expf(d10 - m1), e11 = expf(d11 - m1);
  go[row0 * 256 + cb] = (e00 * v0 + e01 * v1) / (e00 + e01);
  go[row1 * 256 + cb] = (e10 * v0 + e11 * v1) / (e10 + e11);
}

// ---------------------------------------------------------------------------
// Fused cm + cm1 with double-buffered ASYNC global->LDS staging:
// issue bin n+1's tiles while WMMAs consume bin n. h layout
// [br][b][n][1024][128] bf16 -> per-bin 64x128 tiles are contiguous 16 KB.
// ---------------------------------------------------------------------------
__global__ __launch_bounds__(256) void cm_kernel(const __bf16* __restrict__ h,
                                                 const float* __restrict__ agg,
                                                 float* __restrict__ cm_out,
                                                 float* __restrict__ cm1) {
  __shared__ __align__(16) __bf16 s1[2][64 * 128];
  __shared__ __align__(16) __bf16 s2[2][64 * 128];
  const int b = blockIdx.z;
  const int i0 = blockIdx.y * 64, j0 = blockIdx.x * 64;
  const int tid = threadIdx.x, lane = tid & 31, wave = tid >> 5;
  const int wx = wave & 3, wy = wave >> 2;
  const int mS0 = wy * 32, mS1 = mS0 + 16, nS = wx * 16;
  const int half = lane >> 4, col = nS + (lane & 15);
  gf gcm = (gf)cm_out;
  float a0acc[8] = {}, a1acc[8] = {};

  // issue async loads of bin n's two 64x128 tiles into buffer `buf`
  auto issue = [&](int n, int buf) {
    const __bf16* g1 = h + ((size_t)(b * 16 + n)) * 131072 + (size_t)i0 * 128;
    const __bf16* g2 =
        h + ((size_t)((2 + b) * 16 + n)) * 131072 + (size_t)j0 * 128;
    const unsigned l1 = lds_addr(&s1[buf][0]);
    const unsigned l2 = lds_addr(&s2[buf][0]);
#pragma unroll
    for (int r = 0; r < 4; ++r) {
      const int idx = tid + r * 256;  // 16-byte chunk index (1024 total)
      async_ld_b128(l1 + idx * 16, g1 + idx * 8);
      async_ld_b128(l2 + idx * 16, g2 + idx * 8);
    }
  };

  issue(0, 0);
  for (int n = 0; n < 16; ++n) {
    const int buf = n & 1;
    wait_asynccnt0();   // this wave's staged data is in LDS
    __syncthreads();    // every wave's staged data is in LDS
    if (n + 1 < 16) issue(n + 1, buf ^ 1);  // prefetch next bin

    v8f c0 = {}, c1 = {};
#pragma unroll
    for (int kt = 0; kt < 4; ++kt) {
      v16bf bf = frag_from_lds(s2[buf], 128, nS, kt * 32, lane);
      v16bf a0 = frag_from_lds(s1[buf], 128, mS0, kt * 32, lane);
      v16bf a1 = frag_from_lds(s1[buf], 128, mS1, kt * 32, lane);
      c0 = wmma_bf16(a0, bf, c0);
      c1 = wmma_bf16(a1, bf, c1);
    }
    const float agn = ((gcf)agg)[n];
    const size_t base = ((size_t)(b * 16 + n)) * 1024 * 1024;
#pragma unroll
    for (int r = 0; r < 8; ++r) {
      const int m0g = i0 + mS0 + r + 8 * half;
      const int m1g = i0 + mS1 + r + 8 * half;
      gcm[base + (size_t)m0g * 1024 + j0 + col] = c0[r];
      gcm[base + (size_t)m1g * 1024 + j0 + col] = c1[r];
      a0acc[r] += agn * c0[r];
      a1acc[r] += agn * c1[r];
    }
    __syncthreads();  // all waves done reading `buf` before it is re-filled
  }
  gf gc1 = (gf)cm1;
#pragma unroll
  for (int r = 0; r < 8; ++r) {
    const int m0g = i0 + mS0 + r + 8 * half;
    const int m1g = i0 + mS1 + r + 8 * half;
    gc1[((size_t)b * 1024 + m0g) * 1024 + j0 + col] = tanhf(a0acc[r]);
    gc1[((size_t)b * 1024 + m1g) * 1024 + j0 + col] = tanhf(a1acc[r]);
  }
}

// ---------------------------------------------------------------------------
// Per-row windowed sums: rowW[b,r,j] = sum_{c=j}^{j+1014} cm1[b,r,c]
//                                    = total - suffix[9-j] - prefix[j]
// ---------------------------------------------------------------------------
__global__ __launch_bounds__(256) void rowwin_kernel(const float* __restrict__ cm1,
                                                     float* __restrict__ rowW) {
  __shared__ float red[256];
  const int idx = blockIdx.x;  // b*1024 + r
  const gcf row = (gcf)cm1 + (size_t)idx * 1024;
  const int t = threadIdx.x;
  red[t] = row[t] + row[t + 256] + row[t + 512] + row[t + 768];
  __syncthreads();
  for (int off = 128; off > 0; off >>= 1) {
    if (t < off) red[t] += red[t + off];
    __syncthreads();
  }
  if (t == 0) {
    const float total = red[0];
    float pref[10], suf[10];
    pref[0] = 0.f;
    suf[0] = 0.f;
    for (int c = 0; c < 9; ++c) pref[c + 1] = pref[c] + row[c];
    for (int c = 0; c < 9; ++c) suf[c + 1] = suf[c] + row[1023 - c];
    for (int j = 0; j < 10; ++j)
      ((gf)rowW)[(size_t)idx * 10 + j] = total - suf[9 - j] - pref[j];
  }
}

// wsum[b,i,j] = sum_{r=i}^{i+1014} rowW[b,r,j]; pp = sigmoid(tanh(wsum)@Lw + b)
__global__ __launch_bounds__(128) void final_kernel(const float* __restrict__ rowW,
                                                    const float* __restrict__ Lw,
                                                    const float* __restrict__ bsc,
                                                    float* __restrict__ out) {
  __shared__ float red[128];
  const int b = blockIdx.x, t = threadIdx.x;
  float v = 0.f;
  if (t < 100) {
    const int i = t / 10, j = t % 10;
    float s = 0.f;
    const gcf base = (gcf)rowW + ((size_t)b * 1024 + i) * 10 + j;
    for (int r = 0; r < 1015; ++r) s += base[(size_t)r * 10];
    v = tanhf(s) * ((gcf)Lw)[t];
  }
  red[t] = v;
  __syncthreads();
  for (int off = 64; off > 0; off >>= 1) {
    if (t < off) red[t] += red[t + off];
    __syncthreads();
  }
  if (t == 0)
    ((gf)out)[(size_t)2 * 16 * 1024 * 1024 + b] =
        1.f / (1.f + expf(-(red[0] + ((gcf)bsc)[0])));
}

// ---------------------------------------------------------------------------
extern "C" void kernel_launch(void* const* d_in, const int* in_sizes, int n_in,
                              void* d_out, int out_size, void* d_ws,
                              size_t ws_size, hipStream_t stream) {
  (void)in_sizes; (void)n_in; (void)out_size; (void)ws_size;
  const float* x1        = (const float*)d_in[0];
  const float* x2        = (const float*)d_in[1];
  const float* proj_w    = (const float*)d_in[2];
  const float* proj_b    = (const float*)d_in[3];
  const float* in_proj_w = (const float*)d_in[4];
  const float* in_proj_b = (const float*)d_in[5];
  const float* out_w     = (const float*)d_in[6];
  const float* out_b     = (const float*)d_in[7];
  const float* Wbin      = (const float*)d_in[8];
  const float* agg       = (const float*)d_in[9];
  const float* Lw        = (const float*)d_in[10];
  const float* bsc       = (const float*)d_in[11];
  float* out = (float*)d_out;

  char* ws = (char*)d_ws;
  float*  XP   = (float*)ws;  ws += (size_t)4096 * 256 * 4;   // 4 MB
  float*  QKV  = (float*)ws;  ws += (size_t)4096 * 768 * 4;   // 12 MB
  float*  O    = (float*)ws;  ws += (size_t)4096 * 256 * 4;   // 4 MB
  float*  XA   = (float*)ws;  ws += (size_t)4096 * 256 * 4;   // 4 MB
  __bf16* H    = (__bf16*)ws; ws += (size_t)4 * 16 * 1024 * 128 * 2;  // 16 MB
  float*  CM1  = (float*)ws;  ws += (size_t)2 * 1024 * 1024 * 4;      // 8 MB
  float*  ROWW = (float*)ws;                                   // 80 KB

  const dim3 blk(256);
  // proj (shared weights, one launch per branch): (2048 x 1280) @ (1280 x 256)
  gemm_kernel<<<dim3(4, 32), blk, 0, stream>>>(x1, 1280, proj_w, 1280, 1280, 1,
                                               XP, 256, proj_b, nullptr, 0);
  gemm_kernel<<<dim3(4, 32), blk, 0, stream>>>(x2, 1280, proj_w, 1280, 1280, 1,
                                               XP + (size_t)2048 * 256, 256,
                                               proj_b, nullptr, 0);
  // qkv: (4096 x 256) @ (256 x 768)
  gemm_kernel<<<dim3(12, 64), blk, 0, stream>>>(XP, 256, in_proj_w, 256, 256, 1,
                                                QKV, 768, in_proj_b, nullptr, 0);
  // attention over S=2
  attn_kernel<<<dim3(2048), blk, 0, stream>>>(QKV, O);
  // out proj + residual + tanh -> xa
  gemm_kernel<<<dim3(4, 64), blk, 0, stream>>>(O, 256, out_w, 256, 256, 1, XA,
                                               256, out_b, XP, 1);
  // h = tanh(xa @ Wbin[n]) -> bf16, 64 batched GEMMs
  wbin_kernel<<<dim3(2, 16, 64), blk, 0, stream>>>(XA, Wbin, H);
  // cm = h1 @ h2^T (writes 128 MB to d_out) fused with cm1 accumulation
  cm_kernel<<<dim3(16, 16, 2), blk, 0, stream>>>(H, agg, out, CM1);
  // windowed box sums via prefix/suffix trick (no unfold)
  rowwin_kernel<<<dim3(2048), blk, 0, stream>>>(CM1, ROWW);
  final_kernel<<<dim3(2), dim3(128), 0, stream>>>(ROWW, Lw, bsc, out);
}
